// PremulLayer_71167608095133
// MI455X (gfx1250) — matrix operand order repaired and checked by reading, not compiled, not run on hardware
//
#include <hip/hip_runtime.h>
#include <hip/hip_bf16.h>

// ---------------- problem constants ----------------
namespace {
constexpr int T_STEPS = 1024;
constexpr int BATCH   = 64;
constexpr int IN_DIM  = 512;
constexpr int HID     = 512;
constexpr int GATES   = 4 * HID;       // 2048
constexpr int KDIM    = IN_DIM + HID;  // 1024 (A = [x_t | h])
constexpr int NWG     = 8;             // persistent workgroups (co-resident)
constexpr int WGSIZE  = 128;           // 4 waves of 32
constexpr int WAVES   = WGSIZE / 32;
}

typedef __attribute__((ext_vector_type(16))) __bf16 v16bf;
typedef __attribute__((ext_vector_type(8)))  __bf16 v8bf;
typedef __attribute__((ext_vector_type(4)))  __bf16 v4bf;
typedef __attribute__((ext_vector_type(8)))  float  v8f;
typedef __attribute__((ext_vector_type(4)))  int    v4i;

union V16U { v16bf v; v8bf h[2]; };

// gfx1250 async global->LDS path (ASYNCcnt), with portable fallback.
#if defined(__HIP_DEVICE_COMPILE__) && \
    __has_builtin(__builtin_amdgcn_global_load_async_to_lds_b128)
#define USE_ASYNC_LDS 1
#else
#define USE_ASYNC_LDS 0
#endif

__device__ __forceinline__ float sigmoidf(float x) {
  return 1.0f / (1.0f + __expf(-x));
}

#if USE_ASYNC_LDS
__device__ __forceinline__ void async_copy16(const void* gsrc, void* ldst) {
  // builtin signature: (global v4i*, local v4i*, imm offset, imm cpol)
  auto gp = (__attribute__((address_space(1))) v4i*)
                (unsigned long long)(uintptr_t)gsrc;
  auto lp = (__attribute__((address_space(3))) v4i*)
                (unsigned)(uintptr_t)ldst;   // low 32 bits = LDS offset
  __builtin_amdgcn_global_load_async_to_lds_b128(gp, lp, 0, 0);
}
__device__ __forceinline__ void wait_async0() {
#if __has_builtin(__builtin_amdgcn_s_wait_asynccnt)
  __builtin_amdgcn_s_wait_asynccnt(0);
#else
  asm volatile("s_wait_asynccnt 0x0" ::: "memory");
#endif
}
#endif

// ---------------------------------------------------------------------------
// One-shot parallel f32 -> bf16 conversion of all activations [T,B,I].
// Memory-bound: ~200 MB total at 23.3 TB/s, removes all cvt work from the
// sequential recurrence.
// ---------------------------------------------------------------------------
__global__ void lstm_convert_inputs(const float* __restrict__ x,
                                    __bf16* __restrict__ xb, int n4) {
  int idx = blockIdx.x * blockDim.x + threadIdx.x;
  if (idx >= n4) return;
  float4 s = ((const float4*)x)[idx];
  v4bf d;
  d[0] = (__bf16)s.x; d[1] = (__bf16)s.y;
  d[2] = (__bf16)s.z; d[3] = (__bf16)s.w;
  ((v4bf*)xb)[idx] = d;
}

// ---------------------------------------------------------------------------
// Pack Wcat = [W_ih | W_hh] (f32, [2048 x 1024]) into bf16 B-fragment layout:
// fragment(ntile, ktile) = 32 lanes x 16 bf16, 1KB contiguous.
// B (32x16, KxN): lane L -> n = ntile*16 + (L&15); lanes 16-31 hold K+16.
// ---------------------------------------------------------------------------
__global__ void lstm_pack_weights(const float* __restrict__ wih,
                                  const float* __restrict__ whh,
                                  __bf16* __restrict__ wpack) {
  int idx = blockIdx.x * blockDim.x + threadIdx.x;
  if (idx >= GATES * KDIM) return;
  int e    = idx & 15;
  int lane = (idx >> 4) & 31;
  int kt   = (idx >> 9) & 31;
  int nt   = idx >> 14;
  int n = nt * 16 + (lane & 15);
  int k = kt * 32 + ((lane >> 4) << 4) + e;
  float v = (k < IN_DIM) ? wih[n * IN_DIM + k] : whh[n * HID + (k - IN_DIM)];
  wpack[idx] = (__bf16)v;
}

__global__ void lstm_init_state(const float* __restrict__ h0,
                                const float* __restrict__ c0,
                                __bf16* __restrict__ hbuf0,
                                float* __restrict__ cbuf,
                                unsigned* __restrict__ barrier_cnt) {
  int idx = blockIdx.x * blockDim.x + threadIdx.x;
  if (idx < BATCH * HID) {
    hbuf0[idx] = (__bf16)h0[idx];
    cbuf[idx]  = c0[idx];
  }
  if (idx == 0) *barrier_cnt = 0u;
}

// ---------------------------------------------------------------------------
// Persistent fused LSTM. 8 WGs x 4 waves = 32 waves; wave w owns H-columns
// [strip*16, strip*16+16) of all four gates -> cell update fully in-register.
// Per step per wave: 32 ktiles x 4 gates x 4 mtiles = 512 WMMAs.
// ---------------------------------------------------------------------------
__global__ void __launch_bounds__(WGSIZE, 1)
lstm_persistent(const __bf16* __restrict__ xbf,
                const float* __restrict__ bias,
                const __bf16* __restrict__ wpack,
                __bf16* __restrict__ hbuf0,
                __bf16* __restrict__ hbuf1,
                float* __restrict__ cbuf,
                unsigned* __restrict__ barrier_cnt,
                float* __restrict__ out) {
  // A tile [64 x 1024] bf16, row-major: 128 KB of the WGP's 320 KB LDS.
  __shared__ __align__(16) __bf16 Atile[BATCH * KDIM];

  const int tid   = threadIdx.x;
  const int lane  = tid & 31;
  const int wave  = tid >> 5;
  const int strip = blockIdx.x * WAVES + wave;   // 0..31: 16-col strip of H
  const int lrow  = lane & 15;
  const int lhalf = (lane >> 4) << 3;            // A-frag K-half select (0/8)

  __bf16* hb[2] = { hbuf0, hbuf1 };

  float bg[4];
#pragma unroll
  for (int g = 0; g < 4; ++g) bg[g] = bias[g * HID + strip * 16 + lrow];
  const int colg = strip * 16 + lrow;

  for (int t = 0; t < T_STEPS; ++t) {
    const __bf16* __restrict__ hin  = hb[t & 1];
    __bf16* __restrict__       hout = hb[(t + 1) & 1];
    const __bf16* __restrict__ xt   = xbf + (size_t)t * (BATCH * IN_DIM);

    // ---- stage A = [x_t | h] into LDS: pure bf16 byte copy, async path
    for (int i = tid; i < (BATCH * KDIM) / 8; i += WGSIZE) {
      int elem = i * 8;                 // 16B chunk
      int row  = elem >> 10;            // /KDIM
      int col  = elem & (KDIM - 1);
      const __bf16* src = (col < IN_DIM)
                              ? (xt + row * IN_DIM + col)
                              : (hin + row * HID + (col - IN_DIM));
      if (col < IN_DIM && (col & 255) == 0 && (t + 1) < T_STEPS)
        __builtin_prefetch(xt + BATCH * IN_DIM + row * IN_DIM + col, 0, 0);
#if USE_ASYNC_LDS
      async_copy16(src, Atile + elem);  // global_load_async_to_lds_b128
#else
      *(v8bf*)(Atile + elem) = *(const v8bf*)src;
#endif
    }
#if USE_ASYNC_LDS
    wait_async0();                      // s_wait_asynccnt 0
#endif
    __syncthreads();

    // ---- GEMM: gates[g] strip = A(64x1024) x Wcat^T(1024x16), f32 acc
    v8f acc[4][4];
#pragma unroll
    for (int g = 0; g < 4; ++g)
#pragma unroll
      for (int m = 0; m < 4; ++m)
        acc[g][m] = (v8f){0.f,0.f,0.f,0.f,0.f,0.f,0.f,0.f};

    for (int kt = 0; kt < 32; ++kt) {
      V16U a[4];
#pragma unroll
      for (int m = 0; m < 4; ++m) {
        int base = (m * 16 + lrow) * KDIM + kt * 32 + (lhalf << 1);
        a[m].h[0] = *(const v8bf*)(Atile + base);        // ds_load_b128
        a[m].h[1] = *(const v8bf*)(Atile + base + 8);    // ds_load_b128
      }
#pragma unroll
      for (int g = 0; g < 4; ++g) {
        size_t fidx = ((size_t)(g * 32 + strip) * 32 + kt) * 32 + lane;
        v16bf b = *(const v16bf*)(wpack + fidx * 16);    // 32B/lane, L2-hot
#pragma unroll
        for (int m = 0; m < 4; ++m)
          acc[g][m] = __builtin_amdgcn_wmma_f32_16x16x32_bf16(
              false, a[m].v, false, b, (short)0, acc[g][m], false, false);
      }
    }

    // ---- cell update, fully in-register: i/f/z/o share (lane, j) position
#pragma unroll
    for (int m = 0; m < 4; ++m) {
#pragma unroll
      for (int j = 0; j < 8; ++j) {
        int row = m * 16 + j + lhalf;
        float ig = sigmoidf(acc[0][m][j] + bg[0]);
        float fg = sigmoidf(acc[1][m][j] + bg[1]);
        float zg = sigmoidf(acc[2][m][j] + bg[2]);
        float og = sigmoidf(acc[3][m][j] + bg[3]);
        float cold = cbuf[row * HID + colg];         // wave-private columns
        float cnew = fg * cold + (zg - ig);
        float hnew = sigmoidf(cnew) - og;
        cbuf[row * HID + colg] = cnew;
        hout[row * HID + colg] = (__bf16)hnew;
        out[(size_t)t * (BATCH * HID) + row * HID + colg] = hnew;
      }
    }

    // ---- grid-wide barrier: next step's A-stage reads everyone's h
    __syncthreads();
    if (tid == 0) {
      __threadfence();
      __hip_atomic_fetch_add(barrier_cnt, 1u, __ATOMIC_ACQ_REL,
                             __HIP_MEMORY_SCOPE_AGENT);
      const unsigned target = (unsigned)(t + 1) * NWG;
      while (__hip_atomic_load(barrier_cnt, __ATOMIC_ACQUIRE,
                               __HIP_MEMORY_SCOPE_AGENT) < target) {
        __builtin_amdgcn_s_sleep(1);
      }
    }
    __syncthreads();
    __threadfence();
  }

  // ---- tail outputs: h_f = outputs[T-1] (exact f32), c_f from state
  float* hf_out = out + (size_t)T_STEPS * BATCH * HID;
  float* cf_out = hf_out + BATCH * HID;
  const float* hlast = out + (size_t)(T_STEPS - 1) * BATCH * HID;
  for (int idx = blockIdx.x * WGSIZE + tid; idx < BATCH * HID;
       idx += NWG * WGSIZE) {
    hf_out[idx] = hlast[idx];
    cf_out[idx] = cbuf[idx];
  }
}

// ---------------------------------------------------------------------------
extern "C" void kernel_launch(void* const* d_in, const int* in_sizes, int n_in,
                              void* d_out, int out_size, void* d_ws,
                              size_t ws_size, hipStream_t stream) {
  (void)in_sizes; (void)n_in; (void)out_size; (void)ws_size;

  const float* inputs = (const float*)d_in[0];   // [T,B,I]
  const float* h0     = (const float*)d_in[1];   // [B,H]
  const float* c0     = (const float*)d_in[2];   // [B,H]
  const float* wih    = (const float*)d_in[3];   // [4H,I]
  const float* whh    = (const float*)d_in[4];   // [4H,H]
  const float* bias   = (const float*)d_in[5];   // [4H]
  float* out = (float*)d_out;                    // [T,B,H] ++ [B,H] ++ [B,H]

  // workspace layout (~68.5 MB)
  char* ws = (char*)d_ws;
  __bf16*   wpack = (__bf16*)ws;                          // 4 MB
  __bf16*   xbf   = (__bf16*)(ws + (4u << 20));           // 64 MB [T,B,I] bf16
  __bf16*   hbuf0 = (__bf16*)(ws + (68u << 20));          // 64 KB
  __bf16*   hbuf1 = hbuf0 + BATCH * HID;                  // 64 KB
  float*    cbuf  = (float*)(hbuf1 + BATCH * HID);        // 128 KB
  unsigned* bar   = (unsigned*)(cbuf + BATCH * HID);

  const int n4 = (T_STEPS * BATCH * IN_DIM) / 4;
  lstm_convert_inputs<<<(n4 + 255) / 256, 256, 0, stream>>>(inputs, xbf, n4);
  lstm_pack_weights<<<(GATES * KDIM + 255) / 256, 256, 0, stream>>>(wih, whh,
                                                                    wpack);
  lstm_init_state<<<(BATCH * HID + 255) / 256, 256, 0, stream>>>(h0, c0, hbuf0,
                                                                 cbuf, bar);
  lstm_persistent<<<NWG, WGSIZE, 0, stream>>>(xbf, bias, wpack, hbuf0, hbuf1,
                                              cbuf, bar, out);
}
